// Drifting_13950053777833
// MI455X (gfx1250) — matrix operand order repaired and checked by reading, not compiled, not run on hardware
//
#include <hip/hip_runtime.h>
#include <hip/hip_bf16.h>
#include <math.h>

// ---------------------------------------------------------------------------
// Problem constants (fixed by the reference: x (4096,512), y_pos/y_neg (4096,512))
// ---------------------------------------------------------------------------
#define NROWS   4096          // rows of x
#define NPOS    4096          // rows of y_pos
#define NTOT    8192          // NPOS + NNEG
#define DIM     512           // feature dim
#define TEMP_INV 20.0f        // 1 / 0.05
#define MASKV   1000000.0f

typedef __attribute__((ext_vector_type(16))) __bf16 bf16x16;
typedef __attribute__((ext_vector_type(8)))  float  floatx8;

union FragB { bf16x16 v; uint4 q[2]; };

// ---------------------------------------------------------------------------
// Pack fp32 -> split bf16 (hi + lo): bf16 WMMA 3-product reaches ~fp32
// ---------------------------------------------------------------------------
__global__ __launch_bounds__(256) void pack_bf16(const float* __restrict__ src,
                                                 __bf16* __restrict__ hi,
                                                 __bf16* __restrict__ lo, int n) {
    int idx = blockIdx.x * 256 + threadIdx.x;
    if (idx < n) {
        float v = src[idx];
        __bf16 h = (__bf16)v;
        hi[idx] = h;
        lo[idx] = (__bf16)(v - (float)h);
    }
}

// Tiled transpose: y (rowsxDIM fp32, row j) -> Yt[d][jbase+j] split bf16.
// 64x64 tiles through LDS so both global read and write are coalesced.
__global__ __launch_bounds__(256) void pack_y_t(const float* __restrict__ src,
                                                __bf16* __restrict__ thi,
                                                __bf16* __restrict__ tlo,
                                                int jbase) {
    __shared__ float tile[64][65];
    const int jb = blockIdx.x * 64, db = blockIdx.y * 64;
    const int t = threadIdx.x;
    const int c4 = (t & 15) * 4;   // col group of 4
    const int r0 = t >> 4;         // 0..15
    #pragma unroll
    for (int p = 0; p < 4; p++) {
        int r = p * 16 + r0;
        float4 v = *(const float4*)(src + (size_t)(jb + r) * DIM + db + c4);
        tile[r][c4] = v.x; tile[r][c4 + 1] = v.y;
        tile[r][c4 + 2] = v.z; tile[r][c4 + 3] = v.w;
    }
    __syncthreads();
    #pragma unroll
    for (int p = 0; p < 4; p++) {
        int d = p * 16 + r0;
        size_t o = (size_t)(db + d) * NTOT + jbase + jb + c4;
        #pragma unroll
        for (int e = 0; e < 4; e++) {
            float v = tile[c4 + e][d];
            __bf16 h = (__bf16)v;
            thi[o + e] = h;
            tlo[o + e] = (__bf16)(v - (float)h);
        }
    }
}

// Per-row sum of squares (fp32 source for exactness)
__global__ __launch_bounds__(256) void row_sq(const float* __restrict__ src,
                                              float* __restrict__ dst) {
    __shared__ float s[256];
    int row = blockIdx.x, t = threadIdx.x;
    const float* p = src + (size_t)row * DIM;
    float a = p[t], b = p[t + 256];
    s[t] = a * a + b * b;
    __syncthreads();
    for (int st = 128; st > 0; st >>= 1) {
        if (t < st) s[t] += s[t + st];
        __syncthreads();
    }
    if (t == 0) dst[row] = s[0];
}

// ---------------------------------------------------------------------------
// GEMM 1: L[i][j] = -(sqrt(max(|x_i|^2 + |y_j|^2 - 2 x_i.y_j, 0)) + mask)/TEMP
// Block 256 thr (8 waves), tile 128x128. Wave = 64x32 (4x2 C-tiles).
// A = X rows, B = Y rows: both K-contiguous -> direct global b128 fragment
// loads; operands (~24 MB bf16) are L2-resident (192 MB L2).
// ---------------------------------------------------------------------------
__global__ __launch_bounds__(256) void gemm_logits(
    const __bf16* __restrict__ Xhi, const __bf16* __restrict__ Xlo,
    const __bf16* __restrict__ Yhi, const __bf16* __restrict__ Ylo,
    const float* __restrict__ xsq, const float* __restrict__ ysq,
    float* __restrict__ L) {
    const int tid = threadIdx.x;
    const int lane = tid & 31;
    const int wv = tid >> 5;          // 0..7
    const int wm = wv >> 2;           // 0..1  (M)
    const int wn = wv & 3;            // 0..3  (N)
    const int i0 = blockIdx.y * 128 + wm * 64;
    const int j0 = blockIdx.x * 128 + wn * 32;
    const int lr = lane & 15;
    const int hs = lane >> 4;

    floatx8 acc[4][2];
    for (int a = 0; a < 4; a++)
        for (int b = 0; b < 2; b++)
            for (int e = 0; e < 8; e++) acc[a][b][e] = 0.0f;

    for (int d0 = 0; d0 < DIM; d0 += 32) {
        FragB ahi[4], alo[4], bhi[2], blo[2];
        // A fragments: 16x32 bf16; lane holds K chunks {hs*8..+7, hs*8+16..+23}
        #pragma unroll
        for (int mt = 0; mt < 4; mt++) {
            size_t ro = (size_t)(i0 + mt * 16 + lr) * DIM + d0 + hs * 8;
            const uint4* ph = (const uint4*)(Xhi + ro);
            const uint4* pl = (const uint4*)(Xlo + ro);
            ahi[mt].q[0] = ph[0]; ahi[mt].q[1] = ph[2];
            alo[mt].q[0] = pl[0]; alo[mt].q[1] = pl[2];
        }
        // B fragments: 32x16 bf16; lane holds K chunks {hs*8..+7, hs*8+16..+23}
        #pragma unroll
        for (int nt = 0; nt < 2; nt++) {
            size_t ro = (size_t)(j0 + nt * 16 + lr) * DIM + d0 + hs * 8;
            const uint4* ph = (const uint4*)(Yhi + ro);
            const uint4* pl = (const uint4*)(Ylo + ro);
            bhi[nt].q[0] = ph[0]; bhi[nt].q[1] = ph[2];
            blo[nt].q[0] = pl[0]; blo[nt].q[1] = pl[2];
        }
        #pragma unroll
        for (int mt = 0; mt < 4; mt++)
            #pragma unroll
            for (int nt = 0; nt < 2; nt++) {
                acc[mt][nt] = __builtin_amdgcn_wmma_f32_16x16x32_bf16(
                    false, ahi[mt].v, false, bhi[nt].v, (short)0, acc[mt][nt], false, false);
                acc[mt][nt] = __builtin_amdgcn_wmma_f32_16x16x32_bf16(
                    false, ahi[mt].v, false, blo[nt].v, (short)0, acc[mt][nt], false, false);
                acc[mt][nt] = __builtin_amdgcn_wmma_f32_16x16x32_bf16(
                    false, alo[mt].v, false, bhi[nt].v, (short)0, acc[mt][nt], false, false);
            }
    }
    // Epilogue: distance -> logit (diag mask folded in), write L
    for (int mt = 0; mt < 4; mt++)
        for (int nt = 0; nt < 2; nt++) {
            int j = j0 + nt * 16 + lr;
            float ys = ysq[j];
            #pragma unroll
            for (int r = 0; r < 8; r++) {
                int i = i0 + mt * 16 + r + hs * 8;
                float s = acc[mt][nt][r];
                float d2 = xsq[i] + ys - 2.0f * s;
                float d = sqrtf(fmaxf(d2, 0.0f));
                if (j >= NPOS && (j - NPOS) == i) d += MASKV;
                L[(size_t)i * NTOT + j] = -d * TEMP_INV;
            }
        }
}

// ---------------------------------------------------------------------------
// Online row softmax stats: br[i] = -0.5*(rowmax + log(rowsum))
// ---------------------------------------------------------------------------
__global__ __launch_bounds__(256) void row_softmax(const float* __restrict__ L,
                                                   float* __restrict__ br) {
    __shared__ float sm[256], ss[256];
    int i = blockIdx.x, t = threadIdx.x;
    const float* row = L + (size_t)i * NTOT;
    float m = -3.4e38f, s = 0.0f;
    for (int j = t; j < NTOT; j += 256) {
        float v = row[j];
        if (v > m) { s = s * __expf(m - v) + 1.0f; m = v; }
        else       { s += __expf(v - m); }
    }
    sm[t] = m; ss[t] = s; __syncthreads();
    for (int st = 128; st > 0; st >>= 1) {
        if (t < st) {
            float m2 = sm[t + st], s2 = ss[t + st];
            float mm = fmaxf(sm[t], m2);
            ss[t] = ss[t] * __expf(sm[t] - mm) + s2 * __expf(m2 - mm);
            sm[t] = mm;
        }
        __syncthreads();
    }
    if (t == 0) br[i] = -0.5f * (sm[0] + __logf(ss[0]));
}

// Online col softmax stats: bc[j] = -0.5*(colmax + log(colsum)); 64 cols/block
__global__ __launch_bounds__(256) void col_softmax(const float* __restrict__ L,
                                                   float* __restrict__ bc) {
    __shared__ float sm[256], ss[256];
    int t = threadIdx.x;
    int col = blockIdx.x * 64 + (t & 63);
    int part = t >> 6;                     // 0..3
    float m = -3.4e38f, s = 0.0f;
    for (int r = part; r < NROWS; r += 4) {
        float v = L[(size_t)r * NTOT + col];
        if (v > m) { s = s * __expf(m - v) + 1.0f; m = v; }
        else       { s += __expf(v - m); }
    }
    sm[t] = m; ss[t] = s; __syncthreads();
    if (part == 0) {
        float mm = m, sc = s;
        for (int p = 1; p < 4; p++) {
            float m2 = sm[t + p * 64], s2 = ss[t + p * 64];
            float mx = fmaxf(mm, m2);
            sc = sc * __expf(mm - mx) + s2 * __expf(m2 - mx);
            mm = mx;
        }
        bc[col] = -0.5f * (mm + __logf(sc));
    }
}

// Per-row cross-sums of A = exp(L + br[i] + bc[j]):
//   lpos[i] = br[i] + log(sum over NEG cols)   (scales A_pos)
//   lneg[i] = br[i] + log(sum over POS cols)   (scales A_neg)
__global__ __launch_bounds__(256) void row_absums(
    const float* __restrict__ L, const float* __restrict__ br,
    const float* __restrict__ bc, float* __restrict__ lpos,
    float* __restrict__ lneg) {
    __shared__ float sp[256], sn[256];
    int i = blockIdx.x, t = threadIdx.x;
    float b = br[i];
    float ap = 0.0f, an = 0.0f;
    for (int j = t; j < NTOT; j += 256) {
        float a = __expf(L[(size_t)i * NTOT + j] + b + bc[j]);
        if (j < NPOS) ap += a; else an += a;
    }
    sp[t] = ap; sn[t] = an; __syncthreads();
    for (int st = 128; st > 0; st >>= 1) {
        if (t < st) { sp[t] += sp[t + st]; sn[t] += sn[t + st]; }
        __syncthreads();
    }
    if (t == 0) {
        lpos[i] = b + __logf(sn[0]);   // A_pos * sum(A_neg)
        lneg[i] = b + __logf(sp[0]);   // A_neg * sum(A_pos)
    }
}

// ---------------------------------------------------------------------------
// GEMM 2: V[i][d] = sum_j W[i][j] * Ycat[j][d] with
//   W = +exp(L + lpos[i] + bc[j])  (j < NPOS),  -exp(L + lneg[i] + bc[j]) else
// Block 256 thr (8 waves), tile 128(i) x 128(d), K = 8192 in steps of 32.
//  - Each wave owns a 16-row M tile: its lanes compute the W A-fragment
//    directly in registers (float4 L reads + TRANS expf, co-executes with
//    WMMA). No LDS, no redundancy for A.
//  - B comes from pre-transposed Yt[d][j] (bf16 hi/lo, L2-resident): one
//    coalesced 16 KB global->LDS copy per k-step shared by all 8 waves.
//  - Wave N = 128 -> 8 C-tiles (64 acc VGPRs), 24 WMMA per k-step per wave.
// ---------------------------------------------------------------------------
__global__ __launch_bounds__(256) void gemm_v(
    const float* __restrict__ L,
    const __bf16* __restrict__ Ythi, const __bf16* __restrict__ Ytlo,
    const float* __restrict__ bc, const float* __restrict__ lpos,
    const float* __restrict__ lneg, float* __restrict__ V) {
    __shared__ __align__(16) __bf16 sBhi[128 * 40];
    __shared__ __align__(16) __bf16 sBlo[128 * 40];

    const int tid = threadIdx.x, lane = tid & 31, wv = tid >> 5;
    const int i0 = blockIdx.y * 128 + wv * 16;   // wave-private M tile
    const int d0 = blockIdx.x * 128;             // block N tile
    const int lr = lane & 15, hs = lane >> 4;

    const int m = i0 + lr;                       // this lane's W row
    const float lp = lpos[m], ln = lneg[m];
    const float* Lrow = L + (size_t)m * NTOT;

    // staging indices for the B tile copy (128 d-rows x 32 k, hi+lo)
    const int srow = tid >> 1;                   // 0..127
    const int spart = tid & 1;                   // 0..1 (16 k each)

    floatx8 acc[8];
    for (int a = 0; a < 8; a++)
        for (int e = 0; e < 8; e++) acc[a][e] = 0.0f;

    for (int k0 = 0; k0 < NTOT; k0 += 32) {
        // ---- stage B tile: Yt[d0..d0+127][k0..k0+31] -> LDS (coalesced) ----
        {
            size_t go = (size_t)(d0 + srow) * NTOT + k0 + spart * 16;
            const uint4* gh = (const uint4*)(Ythi + go);
            const uint4* gl = (const uint4*)(Ytlo + go);
            uint4* lh = (uint4*)(sBhi + srow * 40 + spart * 16);
            uint4* ll = (uint4*)(sBlo + srow * 40 + spart * 16);
            lh[0] = gh[0]; lh[1] = gh[1];
            ll[0] = gl[0]; ll[1] = gl[1];
        }
        // ---- per-lane W A-fragment: 16 values in WMMA 16x32 layout ----
        const int kA = k0 + hs * 8;              // chunks {kA..+7, kA+16..+23}
        const float scale = (k0 < NPOS) ? lp : ln;
        const float sgn = (k0 < NPOS) ? 1.0f : -1.0f;
        float4 l0 = *(const float4*)(Lrow + kA);
        float4 l1 = *(const float4*)(Lrow + kA + 4);
        float4 l2 = *(const float4*)(Lrow + kA + 16);
        float4 l3 = *(const float4*)(Lrow + kA + 20);
        float4 b0 = *(const float4*)(bc + kA);
        float4 b1 = *(const float4*)(bc + kA + 4);
        float4 b2 = *(const float4*)(bc + kA + 16);
        float4 b3 = *(const float4*)(bc + kA + 20);
        float lv[16] = {l0.x, l0.y, l0.z, l0.w, l1.x, l1.y, l1.z, l1.w,
                        l2.x, l2.y, l2.z, l2.w, l3.x, l3.y, l3.z, l3.w};
        float bv[16] = {b0.x, b0.y, b0.z, b0.w, b1.x, b1.y, b1.z, b1.w,
                        b2.x, b2.y, b2.z, b2.w, b3.x, b3.y, b3.z, b3.w};
        FragB ahi, alo;
        #pragma unroll
        for (int e = 0; e < 16; e++) {
            float w = sgn * __expf(lv[e] + scale + bv[e]);
            __bf16 h = (__bf16)w;
            ahi.v[e] = h;
            alo.v[e] = (__bf16)(w - (float)h);
        }
        __syncthreads();
        // ---- 8 N-tiles x 3 split-precision WMMAs ----
        #pragma unroll
        for (int nt = 0; nt < 8; nt++) {
            int d = nt * 16 + lr;
            FragB bhi, blo;
            const uint4* ph = (const uint4*)(sBhi + d * 40 + hs * 8);
            const uint4* pl = (const uint4*)(sBlo + d * 40 + hs * 8);
            bhi.q[0] = ph[0]; bhi.q[1] = ph[2];
            blo.q[0] = pl[0]; blo.q[1] = pl[2];
            acc[nt] = __builtin_amdgcn_wmma_f32_16x16x32_bf16(
                false, ahi.v, false, bhi.v, (short)0, acc[nt], false, false);
            acc[nt] = __builtin_amdgcn_wmma_f32_16x16x32_bf16(
                false, ahi.v, false, blo.v, (short)0, acc[nt], false, false);
            acc[nt] = __builtin_amdgcn_wmma_f32_16x16x32_bf16(
                false, alo.v, false, bhi.v, (short)0, acc[nt], false, false);
        }
        __syncthreads();
    }
    // Epilogue: write V (fp32)
    for (int nt = 0; nt < 8; nt++) {
        int dcol = d0 + nt * 16 + lr;
        #pragma unroll
        for (int r = 0; r < 8; r++) {
            int i = i0 + r + hs * 8;
            V[(size_t)i * DIM + dcol] = acc[nt][r];
        }
    }
}

// ---------------------------------------------------------------------------
// Host-side launcher
// ---------------------------------------------------------------------------
extern "C" void kernel_launch(void* const* d_in, const int* in_sizes, int n_in,
                              void* d_out, int out_size, void* d_ws, size_t ws_size,
                              hipStream_t stream) {
    (void)in_sizes; (void)n_in; (void)out_size; (void)ws_size;
    const float* x     = (const float*)d_in[0];
    const float* y_pos = (const float*)d_in[1];
    const float* y_neg = (const float*)d_in[2];
    float* V = (float*)d_out;

    char* base = (char*)d_ws;
    size_t off = 0;
    auto alloc = [&](size_t bytes) -> void* {
        off = (off + 255) & ~(size_t)255;
        void* p = base + off;
        off += bytes;
        return p;
    };

    __bf16* Xhi  = (__bf16*)alloc((size_t)NROWS * DIM * 2);
    __bf16* Xlo  = (__bf16*)alloc((size_t)NROWS * DIM * 2);
    __bf16* Yhi  = (__bf16*)alloc((size_t)NTOT * DIM * 2);   // row-major [j][d]
    __bf16* Ylo  = (__bf16*)alloc((size_t)NTOT * DIM * 2);
    __bf16* Ythi = (__bf16*)alloc((size_t)DIM * NTOT * 2);   // transposed [d][j]
    __bf16* Ytlo = (__bf16*)alloc((size_t)DIM * NTOT * 2);
    float*  xsq  = (float*)alloc((size_t)NROWS * 4);
    float*  ysq  = (float*)alloc((size_t)NTOT * 4);
    float*  br   = (float*)alloc((size_t)NROWS * 4);
    float*  bcv  = (float*)alloc((size_t)NTOT * 4);
    float*  lpos = (float*)alloc((size_t)NROWS * 4);
    float*  lneg = (float*)alloc((size_t)NROWS * 4);
    float*  L    = (float*)alloc((size_t)NROWS * NTOT * 4);  // 128 MB

    const int nx = NROWS * DIM, ny = NPOS * DIM;
    pack_bf16<<<(nx + 255) / 256, 256, 0, stream>>>(x, Xhi, Xlo, nx);
    pack_bf16<<<(ny + 255) / 256, 256, 0, stream>>>(y_pos, Yhi, Ylo, ny);
    pack_bf16<<<(ny + 255) / 256, 256, 0, stream>>>(y_neg,
        Yhi + (size_t)NPOS * DIM, Ylo + (size_t)NPOS * DIM, ny);
    pack_y_t<<<dim3(NPOS / 64, DIM / 64), 256, 0, stream>>>(y_pos, Ythi, Ytlo, 0);
    pack_y_t<<<dim3(NPOS / 64, DIM / 64), 256, 0, stream>>>(y_neg, Ythi, Ytlo, NPOS);
    row_sq<<<NROWS, 256, 0, stream>>>(x, xsq);
    row_sq<<<NPOS, 256, 0, stream>>>(y_pos, ysq);
    row_sq<<<NPOS, 256, 0, stream>>>(y_neg, ysq + NPOS);

    gemm_logits<<<dim3(NTOT / 128, NROWS / 128), 256, 0, stream>>>(
        Xhi, Xlo, Yhi, Ylo, xsq, ysq, L);

    row_softmax<<<NROWS, 256, 0, stream>>>(L, br);
    col_softmax<<<NTOT / 64, 256, 0, stream>>>(L, bcv);
    row_absums<<<NROWS, 256, 0, stream>>>(L, br, bcv, lpos, lneg);

    gemm_v<<<dim3(DIM / 128, NROWS / 128), 256, 0, stream>>>(
        L, Ythi, Ytlo, bcv, lpos, lneg, V);
}
